// TransitionDown_6442450944539
// MI455X (gfx1250) — matrix operand order repaired and checked by reading, not compiled, not run on hardware
//
#include <hip/hip_runtime.h>

// ---------------------------------------------------------------------------
// TransitionDown for MI455X (gfx1250, wave32, WMMA)
//   FPS -> kNN(16) -> gather -> [GEMM 262144x128x64 via v_wmma_f32_16x16x32_f16]
//   -> global-batch BN -> ReLU -> maxpool over 16 neighbors
// ---------------------------------------------------------------------------

#define BATCH   8
#define NPTS    8192
#define NPOINT  2048
#define CIN     64
#define COUT    128
#define KNN     16
#define BN_EPS  1e-5f

typedef __attribute__((ext_vector_type(16))) _Float16 v16h;
typedef __attribute__((ext_vector_type(8)))  float    v8f;

// ---------------------------------------------------------------------------
// 1) Farthest point sampling: one 1024-thread block per batch.
//    Distances live in registers (8 points/thread). Per step: min-update,
//    wave32 shfl argmax, then 32-leader LDS reduce. Also emits new_xyz.
// ---------------------------------------------------------------------------
__global__ __launch_bounds__(1024) void fps_kernel(const float* __restrict__ xyz,
                                                   int* __restrict__ cent,
                                                   float* __restrict__ new_xyz)
{
    const int b = blockIdx.x;
    const int t = threadIdx.x;
    constexpr int PTS = NPTS / 1024; // 8
    const float* base = xyz + (size_t)b * NPTS * 3;

    float px[PTS], py[PTS], pz[PTS], dist[PTS];
#pragma unroll
    for (int i = 0; i < PTS; ++i) {
        int pi = i * 1024 + t;
        px[i] = base[pi * 3 + 0];
        py[i] = base[pi * 3 + 1];
        pz[i] = base[pi * 3 + 2];
        dist[i] = 1e10f;
    }

    __shared__ float s_val[32];
    __shared__ int   s_idx[32];
    __shared__ float s_c[3];
    __shared__ int   s_far;
    if (t == 0) s_far = 0;
    __syncthreads();

    for (int it = 0; it < NPOINT; ++it) {
        if (t == 0) {
            int far = s_far;
            cent[b * NPOINT + it] = far;
            float cx = base[far * 3 + 0];
            float cy = base[far * 3 + 1];
            float cz = base[far * 3 + 2];
            s_c[0] = cx; s_c[1] = cy; s_c[2] = cz;
            float* nx = new_xyz + ((size_t)b * NPOINT + it) * 3;
            nx[0] = cx; nx[1] = cy; nx[2] = cz;
        }
        __syncthreads();
        const float cx = s_c[0], cy = s_c[1], cz = s_c[2];

        float best = -1.0f; int bi = 0;
#pragma unroll
        for (int i = 0; i < PTS; ++i) {
            float dx = px[i] - cx, dy = py[i] - cy, dz = pz[i] - cz;
            float d  = dx * dx + dy * dy + dz * dz;
            d = fminf(dist[i], d);
            dist[i] = d;
            int pi = i * 1024 + t;
            if (d > best || (d == best && pi < bi)) { best = d; bi = pi; }
        }
        // wave32 argmax reduce
#pragma unroll
        for (int off = 16; off > 0; off >>= 1) {
            float ov = __shfl_down(best, off, 32);
            int   oi = __shfl_down(bi,   off, 32);
            if (ov > best || (ov == best && oi < bi)) { best = ov; bi = oi; }
        }
        if ((t & 31) == 0) { s_val[t >> 5] = best; s_idx[t >> 5] = bi; }
        __syncthreads();
        if (t < 32) {
            best = s_val[t]; bi = s_idx[t];
#pragma unroll
            for (int off = 16; off > 0; off >>= 1) {
                float ov = __shfl_down(best, off, 32);
                int   oi = __shfl_down(bi,   off, 32);
                if (ov > best || (ov == best && oi < bi)) { best = ov; bi = oi; }
            }
            if (t == 0) s_far = bi;
        }
        __syncthreads();
    }
}

// ---------------------------------------------------------------------------
// 2) kNN(16): one thread per query; candidate points LDS-tiled; register
//    top-16 guarded by cached worst-distance threshold (hot path = 1 compare).
// ---------------------------------------------------------------------------
__global__ __launch_bounds__(128) void knn_kernel(const float* __restrict__ xyz,
                                                  const float* __restrict__ new_xyz,
                                                  int* __restrict__ knn_idx)
{
    const int q = blockIdx.x * 128 + threadIdx.x;  // 0 .. B*NPOINT-1
    const int b = q / NPOINT;
    const float* base = xyz + (size_t)b * NPTS * 3;

    const float qx = new_xyz[q * 3 + 0];
    const float qy = new_xyz[q * 3 + 1];
    const float qz = new_xyz[q * 3 + 2];

    float kd[KNN]; int ki[KNN];
#pragma unroll
    for (int i = 0; i < KNN; ++i) { kd[i] = 3.4e38f; ki[i] = 0; }
    float worst = 3.4e38f;

    __shared__ float sx[128], sy[128], sz[128];
    for (int tile = 0; tile < NPTS; tile += 128) {
        int j = tile + threadIdx.x;
        sx[threadIdx.x] = base[j * 3 + 0];
        sy[threadIdx.x] = base[j * 3 + 1];
        sz[threadIdx.x] = base[j * 3 + 2];
        __syncthreads();
        for (int jj = 0; jj < 128; ++jj) {
            float dx = qx - sx[jj], dy = qy - sy[jj], dz = qz - sz[jj];
            float d  = dx * dx + dy * dy + dz * dz;
            if (d < worst) {
                int pos = KNN - 1;
                while (pos > 0 && kd[pos - 1] > d) {
                    kd[pos] = kd[pos - 1]; ki[pos] = ki[pos - 1]; --pos;
                }
                kd[pos] = d; ki[pos] = tile + jj;
                worst = kd[KNN - 1];
            }
        }
        __syncthreads();
    }
#pragma unroll
    for (int i = 0; i < KNN; ++i) knn_idx[q * KNN + i] = ki[i];
}

// ---------------------------------------------------------------------------
// WMMA fragment helpers (ISA 16-bit 16x32 A layout, 32x16 B layout, wave32)
//   A lane: M = lane&15, half = lane>>4; elem e -> K = e + 8*half        (e<8)
//                                                  K = e + 8 + 8*half   (e>=8)
//   -> two 16x32 fragments cover K=0..63 with four b128 loads each.
//   B lane: N = lane&15; elem e -> K = e + 8*half (+8 if e>=8), same order as A
// ---------------------------------------------------------------------------
__device__ __forceinline__ void load_a_frags(const float* __restrict__ frow,
                                             int half, v16h& a0, v16h& a1)
{
    const float4* f4 = (const float4*)frow;   // feature row: 64 f32, 256B aligned
#pragma unroll
    for (int c = 0; c < 2; ++c) {             // elem groups e=0..7 and e=8..15
        float4 u0 = f4[c * 4 + half * 2 + 0];
        float4 u1 = f4[c * 4 + half * 2 + 1];
        float4 u2 = f4[8 + c * 4 + half * 2 + 0];
        float4 u3 = f4[8 + c * 4 + half * 2 + 1];
        int e = c * 8;
        a0[e + 0] = (_Float16)u0.x; a0[e + 1] = (_Float16)u0.y;
        a0[e + 2] = (_Float16)u0.z; a0[e + 3] = (_Float16)u0.w;
        a0[e + 4] = (_Float16)u1.x; a0[e + 5] = (_Float16)u1.y;
        a0[e + 6] = (_Float16)u1.z; a0[e + 7] = (_Float16)u1.w;
        a1[e + 0] = (_Float16)u2.x; a1[e + 1] = (_Float16)u2.y;
        a1[e + 2] = (_Float16)u2.z; a1[e + 3] = (_Float16)u2.w;
        a1[e + 4] = (_Float16)u3.x; a1[e + 5] = (_Float16)u3.y;
        a1[e + 6] = (_Float16)u3.z; a1[e + 7] = (_Float16)u3.w;
    }
}

__device__ __forceinline__ void load_b_frags(const _Float16* __restrict__ wrow,
                                             int half, v16h& b0, v16h& b1)
{
#pragma unroll
    for (int e = 0; e < 16; ++e) {
        b0[e] = wrow[8 * half + e + (e & 8)];         // K = e + 8*half (+8 if e>=8)
        b1[e] = wrow[32 + 8 * half + e + (e & 8)];    // K = 32 + e + 8*half (+8 if e>=8)
    }
}

// ---------------------------------------------------------------------------
// 3) Pass 1: gather + WMMA GEMM, per-block partial sum / sumsq per channel.
//    Wave w handles point p = blockIdx*8 + w  (its 16 neighbors = one A tile).
//    Deterministic reduction: each wave owns an LDS slot row, no atomics.
// ---------------------------------------------------------------------------
__global__ __launch_bounds__(256) void gemm_stats_kernel(const float* __restrict__ feature,
                                                         const float* __restrict__ W,
                                                         const int* __restrict__ knn_idx,
                                                         float* __restrict__ part_sum,
                                                         float* __restrict__ part_sq)
{
    __shared__ _Float16 s_W[COUT * CIN];        // 16 KB f16 copy of W
    __shared__ float s_wsum[8 * COUT];          // per-wave column sums
    __shared__ float s_wsq[8 * COUT];
    const int tid = threadIdx.x;
    for (int i = tid; i < COUT * CIN; i += 256) s_W[i] = (_Float16)W[i];
    __syncthreads();

    const int wave = tid >> 5, lane = tid & 31;
    const int p    = blockIdx.x * 8 + wave;     // output point 0..16383
    const int b    = p / NPOINT;
    const int row  = lane & 15;                 // neighbor index within point
    const int half = lane >> 4;

    const int nb = knn_idx[p * KNN + row];
    const float* frow = feature + ((size_t)b * NPTS + nb) * CIN;
    v16h a0, a1;
    load_a_frags(frow, half, a0, a1);

#pragma unroll
    for (int t = 0; t < COUT / 16; ++t) {
        const int o = t * 16 + (lane & 15);
        v16h b0, b1;
        load_b_frags(&s_W[o * CIN], half, b0, b1);
        v8f c = {};
        c = __builtin_amdgcn_wmma_f32_16x16x32_f16(false, a0, false, b0,
                                                   (short)0, c, false, false);
        c = __builtin_amdgcn_wmma_f32_16x16x32_f16(false, a1, false, b1,
                                                   (short)0, c, false, false);
        float s = 0.f, sq = 0.f;
#pragma unroll
        for (int i = 0; i < 8; ++i) { s += c[i]; sq += c[i] * c[i]; }
        s  += __shfl_xor(s,  16, 32);           // fold M=8..15 half-lanes
        sq += __shfl_xor(sq, 16, 32);
        if (half == 0) {                        // unique (wave,o) writer
            s_wsum[wave * COUT + o] = s;
            s_wsq [wave * COUT + o] = sq;
        }
    }
    __syncthreads();
    if (tid < COUT) {
        float s = 0.f, sq = 0.f;
#pragma unroll
        for (int w = 0; w < 8; ++w) {
            s  += s_wsum[w * COUT + tid];
            sq += s_wsq [w * COUT + tid];
        }
        part_sum[(size_t)blockIdx.x * COUT + tid] = s;
        part_sq [(size_t)blockIdx.x * COUT + tid] = sq;
    }
}

// ---------------------------------------------------------------------------
// 4) Fold 2048 block partials -> per-channel BN scale/shift.
// ---------------------------------------------------------------------------
__global__ __launch_bounds__(128) void finalize_stats(const float* __restrict__ part_sum,
                                                      const float* __restrict__ part_sq,
                                                      const float* __restrict__ gamma,
                                                      const float* __restrict__ beta,
                                                      float* __restrict__ scale,
                                                      float* __restrict__ shift)
{
    const int c = threadIdx.x;
    float s = 0.f, sq = 0.f;
    for (int i = 0; i < 2048; ++i) {            // coalesced: stride COUT
        s  += part_sum[(size_t)i * COUT + c];
        sq += part_sq [(size_t)i * COUT + c];
    }
    const float inv_cnt = 1.0f / (float)(BATCH * NPOINT * KNN);
    float m  = s * inv_cnt;
    float v  = sq * inv_cnt - m * m;
    float rs = rsqrtf(v + BN_EPS);
    float sc = rs * gamma[c];
    scale[c] = sc;
    shift[c] = beta[c] - m * sc;
}

// ---------------------------------------------------------------------------
// 5) Pass 2: recompute GEMM (cheap), fuse BN + ReLU + maxpool-over-16.
//    Maxpool collapses inside the wave's C fragment (8 regs + shfl_xor 16).
// ---------------------------------------------------------------------------
__global__ __launch_bounds__(256) void gemm_bn_max_kernel(const float* __restrict__ feature,
                                                          const float* __restrict__ W,
                                                          const int* __restrict__ knn_idx,
                                                          const float* __restrict__ scale,
                                                          const float* __restrict__ shift,
                                                          float* __restrict__ out_feat)
{
    __shared__ _Float16 s_W[COUT * CIN];
    const int tid = threadIdx.x;
    for (int i = tid; i < COUT * CIN; i += 256) s_W[i] = (_Float16)W[i];
    __syncthreads();

    const int wave = tid >> 5, lane = tid & 31;
    const int p    = blockIdx.x * 8 + wave;
    const int b    = p / NPOINT;
    const int row  = lane & 15;
    const int half = lane >> 4;

    const int nb = knn_idx[p * KNN + row];
    const float* frow = feature + ((size_t)b * NPTS + nb) * CIN;
    v16h a0, a1;
    load_a_frags(frow, half, a0, a1);

    float* orow = out_feat + (size_t)p * COUT;
#pragma unroll
    for (int t = 0; t < COUT / 16; ++t) {
        const int o = t * 16 + (lane & 15);
        v16h b0, b1;
        load_b_frags(&s_W[o * CIN], half, b0, b1);
        v8f c = {};
        c = __builtin_amdgcn_wmma_f32_16x16x32_f16(false, a0, false, b0,
                                                   (short)0, c, false, false);
        c = __builtin_amdgcn_wmma_f32_16x16x32_f16(false, a1, false, b1,
                                                   (short)0, c, false, false);
        const float sc = scale[o], sh = shift[o];
        float m = -3.4e38f;
#pragma unroll
        for (int i = 0; i < 8; ++i) m = fmaxf(m, fmaf(sc, c[i], sh));
        m = fmaxf(m, __shfl_xor(m, 16, 32));    // fold M=8..15
        m = fmaxf(m, 0.0f);                     // ReLU (commutes with max)
        if (half == 0) orow[o] = m;
    }
}

// ---------------------------------------------------------------------------
// Launch
// ---------------------------------------------------------------------------
extern "C" void kernel_launch(void* const* d_in, const int* in_sizes, int n_in,
                              void* d_out, int out_size, void* d_ws, size_t ws_size,
                              hipStream_t stream)
{
    const float* xyz     = (const float*)d_in[0];   // [B, N, 3]
    const float* feature = (const float*)d_in[1];   // [B, N, 64]
    const float* W       = (const float*)d_in[2];   // [128, 64]
    const float* gamma   = (const float*)d_in[3];   // [128]
    const float* beta    = (const float*)d_in[4];   // [128]
    (void)in_sizes; (void)n_in; (void)out_size; (void)ws_size;

    char* ws = (char*)d_ws;
    int*   cent     = (int*)  (ws);                                     // 64 KB
    int*   knn      = (int*)  (ws + 65536);                             // 1 MB
    float* part_sum = (float*)(ws + 65536 + 1048576);                   // 1 MB
    float* part_sq  = (float*)(ws + 65536 + 2 * 1048576);               // 1 MB
    float* scale    = (float*)(ws + 65536 + 3 * 1048576);               // 512 B
    float* shift    = scale + COUT;                                     // 512 B

    float* out      = (float*)d_out;
    float* new_xyz  = out;                              // [B, NPOINT, 3]
    float* new_feat = out + (size_t)BATCH * NPOINT * 3; // [B, NPOINT, 128]

    fps_kernel        <<<BATCH, 1024, 0, stream>>>(xyz, cent, new_xyz);
    knn_kernel        <<<BATCH * NPOINT / 128, 128, 0, stream>>>(xyz, new_xyz, knn);
    gemm_stats_kernel <<<BATCH * NPOINT / 8, 256, 0, stream>>>(feature, W, knn,
                                                               part_sum, part_sq);
    finalize_stats    <<<1, 128, 0, stream>>>(part_sum, part_sq, gamma, beta,
                                              scale, shift);
    gemm_bn_max_kernel<<<BATCH * NPOINT / 8, 256, 0, stream>>>(feature, W, knn,
                                                               scale, shift, new_feat);
}
